// HFMiMoV2MoE_61546881352283
// MI455X (gfx1250) — matrix operand order repaired and checked by reading, not compiled, not run on hardware
//
#include <hip/hip_runtime.h>
#include <hip/hip_bf16.h>
#include <stdint.h>

// Problem constants (match reference)
#define BB    2
#define SS    2048
#define HH    1024
#define II    512
#define EE    64
#define KK    8
#define GG    8
#define TGTG  4
#define CAP   1024
#define NTOK  (BB*SS)           // 4096
#define MOE_SCALE 2.5f

typedef __bf16 bf16;
typedef __attribute__((ext_vector_type(4)))  __bf16 v4bf;
typedef __attribute__((ext_vector_type(8)))  __bf16 v8bf;
typedef __attribute__((ext_vector_type(16))) __bf16 v16bf;
typedef __attribute__((ext_vector_type(8)))  float  v8f;

// native hardware fp32 -> bf16 (RNE) conversion (v_cvt_pk_bf16_f32)
__device__ __forceinline__ bf16 f2bf(float f) { return (bf16)f; }

// -------------------------------------------------------------------------
// Kernel 0: zero output accumulator + expert counts
// -------------------------------------------------------------------------
__global__ void moe_init(float* __restrict__ out, int* __restrict__ counts) {
  int i = blockIdx.x * blockDim.x + threadIdx.x;
  const int total = NTOK * HH;
  for (int j = i; j < total; j += gridDim.x * blockDim.x) out[j] = 0.0f;
  if (blockIdx.x == 0 && threadIdx.x < EE) counts[threadIdx.x] = 0;
}

// -------------------------------------------------------------------------
// Kernel 1: gating — sigmoid scores, grouped top-k (DeepSeek style)
// -------------------------------------------------------------------------
__global__ __launch_bounds__(64)
void moe_gate(const float* __restrict__ x, const float* __restrict__ gw,
              const float* __restrict__ gb, int* __restrict__ tidx,
              float* __restrict__ tw) {
  __shared__ float xs[HH];
  __shared__ float sc[EE];
  __shared__ float sfc[EE];
  const int t = blockIdx.x;
  const int tid = threadIdx.x;

  for (int i = tid; i < HH; i += 64) xs[i] = x[(size_t)t * HH + i];
  __syncthreads();

  const float* w = gw + (size_t)tid * HH;
  float acc = 0.0f;
  #pragma unroll 8
  for (int i = 0; i < HH; ++i) acc += xs[i] * w[i];
  const float s = 1.0f / (1.0f + __expf(-acc));
  sc[tid]  = s;
  sfc[tid] = s + gb[tid];
  __syncthreads();

  if (tid == 0) {
    float gsc[GG];
    for (int g = 0; g < GG; ++g) {
      float m1 = -1e30f, m2 = -1e30f;
      for (int j = 0; j < EE / GG; ++j) {
        float v = sfc[g * (EE / GG) + j];
        if (v > m1) { m2 = m1; m1 = v; } else if (v > m2) { m2 = v; }
      }
      gsc[g] = m1 + m2;
    }
    bool gsel[GG];
    for (int g = 0; g < GG; ++g) gsel[g] = false;
    for (int q = 0; q < TGTG; ++q) {
      int bi = 0; float bv = -1e30f;
      for (int g = 0; g < GG; ++g)
        if (!gsel[g] && gsc[g] > bv) { bv = gsc[g]; bi = g; }
      gsel[bi] = true;
    }
    bool taken[EE];
    for (int e = 0; e < EE; ++e) taken[e] = false;
    int   isel[KK];
    float wsel[KK];
    float sum = 0.0f;
    for (int q = 0; q < KK; ++q) {
      int bi = 0; float bv = -1e30f;
      for (int e = 0; e < EE; ++e)
        if (gsel[e >> 3] && !taken[e] && sfc[e] > bv) { bv = sfc[e]; bi = e; }
      taken[bi] = true;
      isel[q] = bi;
      wsel[q] = sc[bi];
      sum += sc[bi];
    }
    const float inv = MOE_SCALE / (sum + 1e-20f);
    for (int q = 0; q < KK; ++q) {
      tidx[(size_t)t * KK + q] = isel[q];
      tw[(size_t)t * KK + q]   = wsel[q] * inv;
    }
  }
}

// -------------------------------------------------------------------------
// Kernel 2: capacity dispatch — slot -> (expert, position)
// -------------------------------------------------------------------------
__global__ void moe_dispatch(const int* __restrict__ tidx, int* __restrict__ counts,
                             int* __restrict__ rows) {
  int slot = blockIdx.x * blockDim.x + threadIdx.x;
  if (slot >= NTOK * KK) return;
  int e = tidx[slot];
  int p = atomicAdd(&counts[e], 1);
  if (p < CAP) rows[e * CAP + p] = slot;
}

// -------------------------------------------------------------------------
// WMMA fragment helpers (bf16 16x16x32, wave32 layouts from CDNA5 ISA 7.12.2)
// -------------------------------------------------------------------------
#define LDT 40   // LDS row stride in bf16 elements (80B, 16B aligned)

__device__ __forceinline__ v16bf load_frag_a(const bf16* base, int lane) {
  int m = lane & 15, h = lane >> 4;
  const bf16* p = base + m * LDT + h * 8;
  v8bf lo = *(const v8bf*)(p);
  v8bf hi = *(const v8bf*)(p + 16);
  return __builtin_shufflevector(lo, hi, 0,1,2,3,4,5,6,7,8,9,10,11,12,13,14,15);
}
__device__ __forceinline__ v16bf load_frag_b(const bf16* base, int lane) {
  int n = lane & 15, kh = lane >> 4;
  const bf16* p = base + n * LDT + kh * 8;
  v8bf lo = *(const v8bf*)(p);
  v8bf hi = *(const v8bf*)(p + 16);
  return __builtin_shufflevector(lo, hi, 0,1,2,3,4,5,6,7,8,9,10,11,12,13,14,15);
}
__device__ __forceinline__ v8f wmma_bf16(v16bf a, v16bf b, v8f c) {
  return __builtin_amdgcn_wmma_f32_16x16x32_bf16(false, a, false, b, (short)0, c,
                                                 false, false);
}

// -------------------------------------------------------------------------
// Kernel 3: fused gather + gate/up GEMMs + SiLU*mul -> act[E][CAP][I] (bf16)
// block = 256 thr (8 waves, 4x2), macro-tile 128(M) x 64(N), K-step 32
// double-buffered LDS, 1 barrier/iter. Out-of-range A rows are index-clamped
// (not zeroed): garbage rows only affect discarded output rows.
// -------------------------------------------------------------------------
__global__ __launch_bounds__(256)
void moe_gateup(const float* __restrict__ x, const float* __restrict__ wg,
                const float* __restrict__ wu, const int* __restrict__ counts,
                const int* __restrict__ rows, bf16* __restrict__ act) {
  __shared__ __align__(16) bf16 As[2][128 * LDT];
  __shared__ __align__(16) bf16 Bg[2][64 * LDT];
  __shared__ __align__(16) bf16 Bu[2][64 * LDT];

  const int e   = blockIdx.z;
  const int cnt = min(counts[e], CAP);
  const int m0  = blockIdx.x * 128;
  if (m0 >= cnt) return;
  const int n0  = blockIdx.y * 64;

  const int tid = threadIdx.x, lane = tid & 31, wv = tid >> 5;
  const int wm = wv & 3, wn = wv >> 2;

  v8f ag[2][2] = {{{0}}};
  v8f au[2][2] = {{{0}}};

  const int rstride = tid >> 3;          // 0..31
  const int c4 = (tid & 7) * 4;          // 0..28 step 4
  int tok[4];                            // clamped gather: always valid
  #pragma unroll
  for (int it = 0; it < 4; ++it) {
    int m = min(m0 + it * 32 + rstride, cnt - 1);
    tok[it] = rows[e * CAP + m] >> 3;    // token = slot / K
  }

  const int n  = tid & 63;
  const int kb = (tid >> 6) * 8;         // 0,8,16,24
  const float* pg0 = wg + ((size_t)e * HH) * II + n0 + n;
  const float* pu0 = wu + ((size_t)e * HH) * II + n0 + n;

  auto stage = [&](int buf, int kk0) {
    // A: gathered tokens, fp32 -> bf16, unconditional clamped loads
    float4 av[4];
    #pragma unroll
    for (int it = 0; it < 4; ++it)
      av[it] = *(const float4*)(x + (size_t)tok[it] * HH + kk0 + c4);
    #pragma unroll
    for (int it = 0; it < 4; ++it) {
      int r = it * 32 + rstride;
      v4bf o;
      o[0] = f2bf(av[it].x); o[1] = f2bf(av[it].y);
      o[2] = f2bf(av[it].z); o[3] = f2bf(av[it].w);
      *(v4bf*)(As[buf] + r * LDT + c4) = o;
    }
    // B: weights, transposed [n][k], fp32 -> bf16
    const float* pg = pg0 + (size_t)kk0 * II;
    const float* pu = pu0 + (size_t)kk0 * II;
    __builtin_prefetch(pg + (size_t)32 * II, 0, 1);
    __builtin_prefetch(pu + (size_t)32 * II, 0, 1);
    float gv[8], uv[8];
    #pragma unroll
    for (int j = 0; j < 8; ++j) {
      gv[j] = pg[(size_t)(kb + j) * II];
      uv[j] = pu[(size_t)(kb + j) * II];
    }
    v4bf g0, g1, u0, u1;
    #pragma unroll
    for (int j = 0; j < 4; ++j) {
      g0[j] = f2bf(gv[j]); g1[j] = f2bf(gv[j + 4]);
      u0[j] = f2bf(uv[j]); u1[j] = f2bf(uv[j + 4]);
    }
    *(v4bf*)(Bg[buf] + n * LDT + kb)     = g0;
    *(v4bf*)(Bg[buf] + n * LDT + kb + 4) = g1;
    *(v4bf*)(Bu[buf] + n * LDT + kb)     = u0;
    *(v4bf*)(Bu[buf] + n * LDT + kb + 4) = u1;
  };

  auto compute = [&](int buf) {
    v16bf bgf[2], buf_[2];
    #pragma unroll
    for (int ni = 0; ni < 2; ++ni) {
      bgf[ni]  = load_frag_b(Bg[buf] + (wn * 32 + ni * 16) * LDT, lane);
      buf_[ni] = load_frag_b(Bu[buf] + (wn * 32 + ni * 16) * LDT, lane);
    }
    #pragma unroll
    for (int mi = 0; mi < 2; ++mi) {
      v16bf a = load_frag_a(As[buf] + (wm * 32 + mi * 16) * LDT, lane);
      #pragma unroll
      for (int ni = 0; ni < 2; ++ni) {
        ag[mi][ni] = wmma_bf16(a, bgf[ni],  ag[mi][ni]);
        au[mi][ni] = wmma_bf16(a, buf_[ni], au[mi][ni]);
      }
    }
  };

  stage(0, 0);
  __syncthreads();
  const int nsteps = HH / 32;
  for (int s = 1; s < nsteps; ++s) {
    stage(s & 1, s * 32);
    compute((s - 1) & 1);
    __syncthreads();
  }
  compute((nsteps - 1) & 1);

  // epilogue: act = silu(g) * u
  const int l = lane & 15, h = lane >> 4;
  #pragma unroll
  for (int mi = 0; mi < 2; ++mi) {
    #pragma unroll
    for (int ni = 0; ni < 2; ++ni) {
      const int ncol = n0 + wn * 32 + ni * 16 + l;
      #pragma unroll
      for (int r = 0; r < 8; ++r) {
        int m = m0 + wm * 32 + mi * 16 + r + 8 * h;
        if (m < cnt) {
          float g = ag[mi][ni][r], u = au[mi][ni][r];
          float sg = g / (1.0f + __expf(-g));
          act[((size_t)e * CAP + m) * II + ncol] = f2bf(sg * u);
        }
      }
    }
  }
}

// -------------------------------------------------------------------------
// Kernel 4: down GEMM + fused weighted scatter-combine into out
// A-tile (bf16 act) staged with CDNA5 async copy (ASYNCcnt-tracked,
// no VGPR round-trip), unconditional with clamped rows.
// -------------------------------------------------------------------------
__global__ __launch_bounds__(256)
void moe_down(const bf16* __restrict__ act, const float* __restrict__ wd,
              const int* __restrict__ counts, const int* __restrict__ rows,
              const float* __restrict__ tw, float* __restrict__ out) {
  __shared__ __align__(16) bf16 As[2][128 * LDT];
  __shared__ __align__(16) bf16 Bd[2][64 * LDT];

  const int e   = blockIdx.z;
  const int cnt = min(counts[e], CAP);
  const int m0  = blockIdx.x * 128;
  if (m0 >= cnt) return;
  const int n0  = blockIdx.y * 64;

  const int tid = threadIdx.x, lane = tid & 31, wv = tid >> 5;
  const int wm = wv & 3, wn = wv >> 2;

  v8f acc[2][2] = {{{0}}};

  const int n  = tid & 63;
  const int kb = (tid >> 6) * 8;
  const float* pd0 = wd + ((size_t)e * II) * HH + n0 + n;

  // per-thread A chunk geometry (two 16B chunks per thread per tile)
  int arow[2], acb[2], amrow[2];
  #pragma unroll
  for (int it = 0; it < 2; ++it) {
    int c = tid + 256 * it;        // chunk 0..511
    arow[it] = c >> 2;             // row 0..127
    acb[it]  = (c & 3) * 8;        // bf16 element offset (16B chunks)
    amrow[it] = min(m0 + arow[it], cnt - 1);
  }

  auto stage = [&](int buf, int kk0) {
    // ---- A from act: async global->LDS, unconditional ----
    #pragma unroll
    for (int it = 0; it < 2; ++it) {
      uint32_t loff = (uint32_t)(uintptr_t)&As[buf][arow[it] * LDT + acb[it]];
      uint64_t ga   = (uint64_t)(uintptr_t)
          (act + ((size_t)e * CAP + amrow[it]) * II + kk0 + acb[it]);
      asm volatile("global_load_async_to_lds_b128 %0, %1, off"
                   :: "v"(loff), "v"(ga) : "memory");
    }
    // ---- B: w_down fp32 -> bf16, transposed [n][k] ----
    const float* pd = pd0 + (size_t)kk0 * HH;
    __builtin_prefetch(pd + (size_t)32 * HH, 0, 1);
    float dv[8];
    #pragma unroll
    for (int j = 0; j < 8; ++j) dv[j] = pd[(size_t)(kb + j) * HH];
    v4bf d0, d1;
    #pragma unroll
    for (int j = 0; j < 4; ++j) { d0[j] = f2bf(dv[j]); d1[j] = f2bf(dv[j + 4]); }
    *(v4bf*)(Bd[buf] + n * LDT + kb)     = d0;
    *(v4bf*)(Bd[buf] + n * LDT + kb + 4) = d1;
  };

  auto compute = [&](int buf) {
    v16bf bf_[2];
    #pragma unroll
    for (int ni = 0; ni < 2; ++ni)
      bf_[ni] = load_frag_b(Bd[buf] + (wn * 32 + ni * 16) * LDT, lane);
    #pragma unroll
    for (int mi = 0; mi < 2; ++mi) {
      v16bf a = load_frag_a(As[buf] + (wm * 32 + mi * 16) * LDT, lane);
      #pragma unroll
      for (int ni = 0; ni < 2; ++ni)
        acc[mi][ni] = wmma_bf16(a, bf_[ni], acc[mi][ni]);
    }
  };

  stage(0, 0);
  asm volatile("s_wait_asynccnt 0x0" ::: "memory");
  __syncthreads();
  const int nsteps = II / 32;
  for (int s = 1; s < nsteps; ++s) {
    stage(s & 1, s * 32);
    compute((s - 1) & 1);
    asm volatile("s_wait_asynccnt 0x0" ::: "memory");
    __syncthreads();
  }
  compute((nsteps - 1) & 1);

  // epilogue: weighted atomic scatter into out[token]
  const int l = lane & 15, h = lane >> 4;
  #pragma unroll
  for (int mi = 0; mi < 2; ++mi) {
    #pragma unroll
    for (int ni = 0; ni < 2; ++ni) {
      const int ncol = n0 + wn * 32 + ni * 16 + l;
      #pragma unroll
      for (int r = 0; r < 8; ++r) {
        int m = m0 + wm * 32 + mi * 16 + r + 8 * h;
        if (m < cnt) {
          int slot = rows[e * CAP + m];
          int token = slot >> 3;              // slot / K
          float w = tw[slot];
          __hip_atomic_fetch_add(&out[(size_t)token * HH + ncol],
                                 acc[mi][ni][r] * w,
                                 __ATOMIC_RELAXED, __HIP_MEMORY_SCOPE_AGENT);
        }
      }
    }
  }
}

// -------------------------------------------------------------------------
// Launch
// -------------------------------------------------------------------------
extern "C" void kernel_launch(void* const* d_in, const int* in_sizes, int n_in,
                              void* d_out, int out_size, void* d_ws, size_t ws_size,
                              hipStream_t stream) {
  const float* x  = (const float*)d_in[0];   // hidden_states [B,S,H]
  const float* gw = (const float*)d_in[1];   // gate_w  [E,H]
  const float* gb = (const float*)d_in[2];   // gate_bias [E]
  const float* wg = (const float*)d_in[3];   // w_gate  [E,H,I]
  const float* wu = (const float*)d_in[4];   // w_up    [E,H,I]
  const float* wd = (const float*)d_in[5];   // w_down  [E,I,H]
  float* out = (float*)d_out;                // [B,S,H] fp32

  char* ws = (char*)d_ws;
  constexpr size_t o_counts = 0;
  constexpr size_t o_rows   = 256;
  constexpr size_t o_tidx   = o_rows + (size_t)EE * CAP * 4;
  constexpr size_t o_tw     = o_tidx + (size_t)NTOK * KK * 4;
  constexpr size_t o_act    = (o_tw + (size_t)NTOK * KK * 4 + 255) & ~(size_t)255;
  int*   counts = (int*)(ws + o_counts);
  int*   rows   = (int*)(ws + o_rows);
  int*   tidx   = (int*)(ws + o_tidx);
  float* tw     = (float*)(ws + o_tw);
  bf16*  act    = (bf16*)(ws + o_act);       // [E][CAP][I] bf16 = 64 MB

  moe_init<<<1024, 256, 0, stream>>>(out, counts);
  moe_gate<<<NTOK, 64, 0, stream>>>(x, gw, gb, tidx, tw);
  moe_dispatch<<<(NTOK * KK + 255) / 256, 256, 0, stream>>>(tidx, counts, rows);

  dim3 g1(CAP / 128, II / 64, EE);   // 8 x 8 x 64
  moe_gateup<<<g1, 256, 0, stream>>>(x, wg, wu, counts, rows, act);

  dim3 g2(CAP / 128, HH / 64, EE);   // 8 x 16 x 64
  moe_down<<<g2, 256, 0, stream>>>(act, wd, counts, rows, tw, out);
}
